// BertSelfAttention_50302656970873
// MI455X (gfx1250) — matrix operand order repaired and checked by reading, not compiled, not run on hardware
//
#include <hip/hip_runtime.h>

typedef __attribute__((ext_vector_type(16))) __bf16       v16bf;
typedef __attribute__((ext_vector_type(8)))  float        v8f;
typedef __attribute__((ext_vector_type(4)))  unsigned int v4u;

union FragB { v4u q[2]; v16bf v; };

__device__ __forceinline__ unsigned short f2bf(float f) {
  unsigned u = __float_as_uint(f);
  u += 0x7FFFu + ((u >> 16) & 1u);            // round-to-nearest-even
  return (unsigned short)(u >> 16);
}

#define BB 2
#define SS 2048
#define DD 1024
#define HH 16
#define HD 64

// ---------------- conversion kernels ----------------
__global__ void cvt_x(const float* __restrict__ X, unsigned short* __restrict__ Xbf, int n) {
  int i = blockIdx.x * blockDim.x + threadIdx.x;
  int stride = gridDim.x * blockDim.x;
  for (; i < n; i += stride) Xbf[i] = f2bf(X[i]);
}

// Wt layout: [3][N=1024][K=1024], Wt[w][n][k] = W[w][k][n]  (bf16)
__global__ void cvt_w(const float* __restrict__ Wq, const float* __restrict__ Wk,
                      const float* __restrict__ Wv, unsigned short* __restrict__ Wt) {
  int i = blockIdx.x * blockDim.x + threadIdx.x;
  int stride = gridDim.x * blockDim.x;
  const int total = 3 * DD * DD;
  for (; i < total; i += stride) {
    int w = i >> 20;
    int rem = i & 0xFFFFF;
    int nn = rem >> 10;
    int kk = rem & (DD - 1);
    const float* W = (w == 0) ? Wq : ((w == 1) ? Wk : Wv);
    Wt[i] = f2bf(W[kk * DD + nn]);
  }
}

// ---------------- head-gate predictor ----------------
__global__ void gate_kernel(const float* __restrict__ X,
                            const float* __restrict__ pW1, const float* __restrict__ pb1,
                            const float* __restrict__ gma, const float* __restrict__ bta,
                            const float* __restrict__ mean, const float* __restrict__ var,
                            const float* __restrict__ pW2, const float* __restrict__ pb2,
                            float* __restrict__ gates) {
  __shared__ float pooled[DD];
  __shared__ float hbuf[64];
  int b = blockIdx.x;
  int t = threadIdx.x;                          // 256 threads
  for (int d = t; d < DD; d += 256) {
    float s = 0.f;
    const float* p = X + (size_t)b * SS * DD + d;
    for (int si = 0; si < SS; ++si) s += p[(size_t)si * DD];
    pooled[d] = s * (1.0f / (float)SS);
  }
  __syncthreads();
  if (t < 64) {
    float hv = pb1[t];
    for (int d = 0; d < DD; ++d) hv += pooled[d] * pW1[d * 64 + t];
    hv = (hv - mean[t]) * rsqrtf(var[t] + 1e-12f) * gma[t] + bta[t];
    hbuf[t] = hv > 0.f ? hv : 0.f;
  }
  __syncthreads();
  if (t < HH) {
    float lg = pb2[t];
    for (int j = 0; j < 64; ++j) lg += hbuf[j] * pW2[j * HH + t];
    gates[b * HH + t] = (lg >= 0.f) ? 1.0f : 0.0f;
  }
}

// ---------------- QKV projection GEMM (WMMA bf16, 64x64 per wave) ----------------
// waves: [3][M/64=64][N/64=16]; each wave -> 4x4 grid of 16x16 WMMA tiles.
__global__ void __launch_bounds__(128) qkv_gemm(
    const unsigned short* __restrict__ Xbf,   // [B*S][D] bf16
    const unsigned short* __restrict__ Wt,    // [3][N][K] bf16 (transposed)
    const float* __restrict__ bq, const float* __restrict__ bk, const float* __restrict__ bv,
    unsigned short* __restrict__ Qbf,         // [B*S][D] bf16
    unsigned short* __restrict__ Kbf,         // [B*S][D] bf16
    unsigned short* __restrict__ Vt) {        // [B*D][S] bf16 (transposed)
  int wave = (blockIdx.x * blockDim.x + threadIdx.x) >> 5;
  int lane = threadIdx.x & 31;
  int which = wave >> 10;                      // / (64*16)
  int rem = wave & 1023;
  int m0 = (rem >> 4) << 6;                    // 64-row tile
  int n0 = (rem & 15) << 6;                    // 64-col tile
  int l15 = lane & 15, kh = lane >> 4;

  const unsigned short* arow[4];
  const unsigned short* brow[4];
  const unsigned short* Wb = Wt + (size_t)which * (DD * DD);
#pragma unroll
  for (int i = 0; i < 4; ++i) {
    arow[i] = Xbf + (size_t)(m0 + i * 16 + l15) * DD;
    brow[i] = Wb + (size_t)(n0 + i * 16 + l15) * DD;
  }

  v8f acc[4][4];
#pragma unroll
  for (int i = 0; i < 4; ++i)
#pragma unroll
    for (int j = 0; j < 4; ++j) acc[i][j] = (v8f){};

  for (int k0 = 0; k0 < DD; k0 += 32) {
    FragB a[4], bfr[4];
#pragma unroll
    for (int i = 0; i < 4; ++i) {
      a[i].q[0] = *(const v4u*)(arow[i] + k0 + kh * 8);        // A: chunks {8kh..},{16+8kh..}
      a[i].q[1] = *(const v4u*)(arow[i] + k0 + 16 + kh * 8);
      bfr[i].q[0] = *(const v4u*)(brow[i] + k0 + kh * 8);      // B: chunks {8kh..},{16+8kh..}
      bfr[i].q[1] = *(const v4u*)(brow[i] + k0 + 16 + kh * 8);
    }
#pragma unroll
    for (int i = 0; i < 4; ++i)
#pragma unroll
      for (int j = 0; j < 4; ++j)
        acc[i][j] = __builtin_amdgcn_wmma_f32_16x16x32_bf16(false, a[i].v, false, bfr[j].v,
                                                            (short)0, acc[i][j], false, false);
  }

  const float* bias = (which == 0) ? bq : ((which == 1) ? bk : bv);
  int n = l15, hi = kh;
  if (which == 2) {
    int b = m0 >> 11;                                    // /2048
#pragma unroll
    for (int j = 0; j < 4; ++j) {
      float bval = bias[n0 + j * 16 + n];
      unsigned short* vcol = Vt + (size_t)(b * DD + n0 + j * 16 + n) * SS;
#pragma unroll
      for (int i = 0; i < 4; ++i) {
#pragma unroll
        for (int r = 0; r < 8; ++r) {
          int m = i * 16 + r + 8 * hi;
          int s = (m0 + m) - (b << 11);
          vcol[s] = f2bf(acc[i][j][r] + bval);
        }
      }
    }
  } else {
    unsigned short* O = (which == 0) ? Qbf : Kbf;
#pragma unroll
    for (int i = 0; i < 4; ++i) {
#pragma unroll
      for (int j = 0; j < 4; ++j) {
        float bval = bias[n0 + j * 16 + n];
#pragma unroll
        for (int r = 0; r < 8; ++r) {
          int m = i * 16 + r + 8 * hi;
          O[(size_t)(m0 + m) * DD + n0 + j * 16 + n] = f2bf(acc[i][j][r] + bval);
        }
      }
    }
  }
}

// ---------------- flash attention (WMMA bf16, online softmax, 32q per wave) ----------------
// waves: [B][H][S/32=64]; each wave owns a 32-query tile, loops 32 keys at a time.
__global__ void __launch_bounds__(128) attn_kernel(
    const unsigned short* __restrict__ Qbf,
    const unsigned short* __restrict__ Kbf,
    const unsigned short* __restrict__ Vt,
    const float* __restrict__ mask,            // [B,1,1,S]
    const float* __restrict__ gates,           // [B,H]
    float* __restrict__ out) {                 // [B,S,D] fp32
  __shared__ __align__(16) unsigned short plds[4][2][16 * 32];
  int wid = threadIdx.x >> 5;
  int lane = threadIdx.x & 31;
  int wave = (blockIdx.x << 2) + wid;          // 2048 waves
  int b = wave >> 10;
  int rem = wave & 1023;
  int h = rem >> 6;
  int m0 = (rem & 63) << 5;                    // 32-query tile
  int l15 = lane & 15, kh = lane >> 4, hi = kh;
  int n = l15;
  const float scale = 0.125f;                  // 1/sqrt(64)

  // Q fragments: 2 query sub-tiles x 2 K-chunks
  FragB aq[2][2];
#pragma unroll
  for (int qt = 0; qt < 2; ++qt) {
    const unsigned short* qrow = Qbf + (size_t)(b * SS + m0 + qt * 16 + l15) * DD + h * HD;
    aq[qt][0].q[0] = *(const v4u*)(qrow + kh * 8);
    aq[qt][0].q[1] = *(const v4u*)(qrow + 16 + kh * 8);
    aq[qt][1].q[0] = *(const v4u*)(qrow + 32 + kh * 8);
    aq[qt][1].q[1] = *(const v4u*)(qrow + 48 + kh * 8);
  }

  v8f ctx[2][4];
  float mrow[2][8], lrow[2][8];
#pragma unroll
  for (int qt = 0; qt < 2; ++qt) {
#pragma unroll
    for (int t = 0; t < 4; ++t) ctx[qt][t] = (v8f){};
#pragma unroll
    for (int r = 0; r < 8; ++r) { mrow[qt][r] = -__builtin_huge_valf(); lrow[qt][r] = 0.f; }
  }

  const float* mbase = mask + (size_t)b * SS;
  const unsigned short* vcol = Vt + (size_t)(b * DD + h * HD + n) * SS;

  for (int j0 = 0; j0 < SS; j0 += 32) {
    // ---- scores: 2 key halves x 2 query sub-tiles; K frags loaded once ----
    v8f sc[2][2];                              // [qt][keyhalf]
#pragma unroll
    for (int kl = 0; kl < 2; ++kl) {
      const unsigned short* krow = Kbf + (size_t)(b * SS + j0 + kl * 16 + n) * DD + h * HD;
      FragB bkA, bkB;
      bkA.q[0] = *(const v4u*)(krow + kh * 8);
      bkA.q[1] = *(const v4u*)(krow + 16 + kh * 8);
      bkB.q[0] = *(const v4u*)(krow + 32 + kh * 8);
      bkB.q[1] = *(const v4u*)(krow + 48 + kh * 8);
      __builtin_prefetch(krow + 32 * DD, 0, 3);          // next key block (global_prefetch_b8)
#pragma unroll
      for (int qt = 0; qt < 2; ++qt) {
        v8f s = {};
        s = __builtin_amdgcn_wmma_f32_16x16x32_bf16(false, aq[qt][0].v, false, bkA.v,
                                                    (short)0, s, false, false);
        s = __builtin_amdgcn_wmma_f32_16x16x32_bf16(false, aq[qt][1].v, false, bkB.v,
                                                    (short)0, s, false, false);
        sc[qt][kl] = s;
      }
    }

    float mlo = mbase[j0 + n], mhi = mbase[j0 + 16 + n];

    // ---- online softmax per query sub-tile ----
#pragma unroll
    for (int qt = 0; qt < 2; ++qt) {
      unsigned short* pb = plds[wid][qt];
#pragma unroll
      for (int r = 0; r < 8; ++r) {
        float sl = sc[qt][0][r] * scale + mlo;
        float sh = sc[qt][1][r] * scale + mhi;
        float mx = fmaxf(sl, sh);
        mx = fmaxf(mx, __shfl_xor(mx, 1, 32));
        mx = fmaxf(mx, __shfl_xor(mx, 2, 32));
        mx = fmaxf(mx, __shfl_xor(mx, 4, 32));
        mx = fmaxf(mx, __shfl_xor(mx, 8, 32));
        float mnew = fmaxf(mrow[qt][r], mx);
        float alpha = __expf(mrow[qt][r] - mnew);
        float pl = __expf(sl - mnew);
        float ph = __expf(sh - mnew);
        float ssum = pl + ph;
        ssum += __shfl_xor(ssum, 1, 32);
        ssum += __shfl_xor(ssum, 2, 32);
        ssum += __shfl_xor(ssum, 4, 32);
        ssum += __shfl_xor(ssum, 8, 32);
        lrow[qt][r] = lrow[qt][r] * alpha + ssum;
        mrow[qt][r] = mnew;
        ctx[qt][0][r] *= alpha; ctx[qt][1][r] *= alpha;
        ctx[qt][2][r] *= alpha; ctx[qt][3][r] *= alpha;
        int rr = r + 8 * hi;
        pb[rr * 32 + n]      = f2bf(pl);
        pb[rr * 32 + 16 + n] = f2bf(ph);
      }
    }
    asm volatile("s_wait_dscnt 0" ::: "memory");

    FragB pf[2];
#pragma unroll
    for (int qt = 0; qt < 2; ++qt) {
      unsigned short* pb = plds[wid][qt];
      pf[qt].q[0] = *(const v4u*)(pb + l15 * 32 + kh * 8);
      pf[qt].q[1] = *(const v4u*)(pb + l15 * 32 + 16 + kh * 8);
    }

    // ---- ctx += P(16x32) x V(32x64): V frags loaded once, used by both sub-tiles ----
#pragma unroll
    for (int t = 0; t < 4; ++t) {
      const unsigned short* vb = vcol + (size_t)(t * 16) * SS + j0;
      FragB bv;
      bv.q[0] = *(const v4u*)(vb + kh * 8);
      bv.q[1] = *(const v4u*)(vb + 16 + kh * 8);
      __builtin_prefetch(vb + 32, 0, 3);                 // next key block of V
      ctx[0][t] = __builtin_amdgcn_wmma_f32_16x16x32_bf16(false, pf[0].v, false, bv.v,
                                                          (short)0, ctx[0][t], false, false);
      ctx[1][t] = __builtin_amdgcn_wmma_f32_16x16x32_bf16(false, pf[1].v, false, bv.v,
                                                          (short)0, ctx[1][t], false, false);
    }
  }

  float gate = gates[b * HH + h];
#pragma unroll
  for (int qt = 0; qt < 2; ++qt) {
#pragma unroll
    for (int r = 0; r < 8; ++r) {
      int m = qt * 16 + r + 8 * hi;
      float inv = gate / lrow[qt][r];
      float* orow = out + (size_t)(b * SS + m0 + m) * DD + h * HD;
      orow[n]      = ctx[qt][0][r] * inv;
      orow[16 + n] = ctx[qt][1][r] * inv;
      orow[32 + n] = ctx[qt][2][r] * inv;
      orow[48 + n] = ctx[qt][3][r] * inv;
    }
  }
}

// ---------------- host launch ----------------
extern "C" void kernel_launch(void* const* d_in, const int* in_sizes, int n_in,
                              void* d_out, int out_size, void* d_ws, size_t ws_size,
                              hipStream_t stream) {
  const float* X    = (const float*)d_in[0];
  const float* mask = (const float*)d_in[1];
  const float* Wq = (const float*)d_in[2];  const float* bq = (const float*)d_in[3];
  const float* Wk = (const float*)d_in[4];  const float* bk = (const float*)d_in[5];
  const float* Wv = (const float*)d_in[6];  const float* bv = (const float*)d_in[7];
  const float* pW1 = (const float*)d_in[8]; const float* pb1 = (const float*)d_in[9];
  const float* gma = (const float*)d_in[10]; const float* bta = (const float*)d_in[11];
  const float* mean = (const float*)d_in[12]; const float* var = (const float*)d_in[13];
  const float* pW2 = (const float*)d_in[14]; const float* pb2 = (const float*)d_in[15];
  float* out = (float*)d_out;
  char* ws = (char*)d_ws;

  // workspace layout (bytes)
  unsigned short* Xbf = (unsigned short*)(ws + 0);          //  8,388,608
  unsigned short* Wt  = (unsigned short*)(ws + 8388608);    //  6,291,456 (3 weights)
  unsigned short* Qbf = (unsigned short*)(ws + 14680064);   //  8,388,608
  unsigned short* Kbf = (unsigned short*)(ws + 23068672);   //  8,388,608
  unsigned short* Vt  = (unsigned short*)(ws + 31457280);   //  8,388,608
  float* gates        = (float*)(ws + 39845888);            //  128

  cvt_x<<<4096, 256, 0, stream>>>(X, Xbf, BB * SS * DD);
  cvt_w<<<4096, 256, 0, stream>>>(Wq, Wk, Wv, Wt);
  gate_kernel<<<BB, 256, 0, stream>>>(X, pW1, pb1, gma, bta, mean, var, pW2, pb2, gates);
  qkv_gemm<<<768, 128, 0, stream>>>(Xbf, Wt, bq, bk, bv, Qbf, Kbf, Vt);
  attn_kernel<<<512, 128, 0, stream>>>(Qbf, Kbf, Vt, mask, gates, out);
}
